// ViTND_16527034155402
// MI455X (gfx1250) — matrix operand order repaired and checked by reading, not compiled, not run on hardware
//
#include <hip/hip_runtime.h>

// ---------------- constants ----------------
#define B_IMG   8
#define NTOK    1024      // tokens per image (32x32 patches)
#define MTOK    8192      // B_IMG * NTOK
#define DIM_    384
#define HEADS_  6
#define DH_     64
#define NF_     32        // DH/2 rotary freqs
#define MLP_    1536
#define NCLS_   1000
#define KPATCH  768       // 16*16*3

typedef __attribute__((ext_vector_type(16))) __bf16 v16bf;
typedef __attribute__((ext_vector_type(8)))  float  v8f;
typedef __attribute__((ext_vector_type(4)))  int    v4i;

union AFrag { v16bf v; unsigned int u[8]; };

#if defined(__has_builtin)
#if __has_builtin(__builtin_amdgcn_global_load_async_to_lds_b128)
#define HAVE_ASYNC_LDS 1
#endif
#endif

#ifdef HAVE_ASYNC_LDS
typedef __attribute__((address_space(1))) v4i* v4i_gptr;
typedef __attribute__((address_space(3))) v4i* v4i_lptr;
#endif

static __device__ __forceinline__ unsigned short f2bf(float f) {
    unsigned int u = __float_as_uint(f);
    u = u + 0x7FFFu + ((u >> 16) & 1u);       // round to nearest even
    return (unsigned short)(u >> 16);
}

// copy 16 bytes global -> LDS (async CDNA5 path if available)
static __device__ __forceinline__ void copy16_g2l(const unsigned short* g, unsigned short* l) {
#ifdef HAVE_ASYNC_LDS
    __builtin_amdgcn_global_load_async_to_lds_b128(
        (v4i_gptr)(v4i*)g, (v4i_lptr)(v4i*)l, 0, 0);
#else
    *(uint4*)l = *(const uint4*)g;
#endif
}
static __device__ __forceinline__ void wait_async_lds() {
#ifdef HAVE_ASYNC_LDS
#if __has_builtin(__builtin_amdgcn_s_wait_asynccnt)
    __builtin_amdgcn_s_wait_asynccnt(0);
#else
    asm volatile("s_wait_asynccnt 0x0" ::: "memory");
#endif
#endif
}

// ---------------- generic bf16 WMMA GEMM ----------------
// C[M,N] = A[M,K](bf16,row-major,lda) * B^T   where B is stored [N,K] (ldb).
// Requirements: K % 32 == 0, lda/ldb % 8 == 0, A/B bases 16B aligned.
// GUARD=false additionally requires M % 64 == 0 and N % 128 == 0.
// Epilogue: +bias, exact GELU (act==1), +res (fp32), out fp32 / bf16 / bf16-transposed.
// Batched over blockIdx.z with strides sA/sB/sC (elements).
template <bool GUARD>
__global__ __launch_bounds__(128)
void wmma_gemm_kernel(const unsigned short* __restrict__ A,
                      const unsigned short* __restrict__ Bg,
                      const float* __restrict__ bias,
                      const float* __restrict__ res,
                      float* __restrict__ outf,
                      unsigned short* __restrict__ outb,
                      unsigned short* __restrict__ outbT,
                      int M, int N, int K,
                      int lda, int ldb, int ldc, int ldcT,
                      int act,
                      long long sA, long long sB, long long sC)
{
    __shared__ __align__(16) unsigned short Asub[64  * 32];   // [m][k]
    __shared__ __align__(16) unsigned short Bsub[128 * 32];   // [n][k]

    const int z = blockIdx.z;
    A  += (long long)z * sA;
    Bg += (long long)z * sB;
    const long long coff = (long long)z * sC;

    const int tid  = threadIdx.x;
    const int m0   = blockIdx.y * 64;
    const int n0   = blockIdx.x * 128;
    const int lane = tid & 31;
    const int wid  = tid >> 5;
    const int wm   = (wid >> 1) * 32;        // wave M offset in block tile
    const int wn   = (wid & 1)  * 64;        // wave N offset in block tile
    const int mrow = lane & 15;
    const int khalf = lane >> 4;

    v8f c[2][4];
#pragma unroll
    for (int mt = 0; mt < 2; ++mt)
#pragma unroll
        for (int nt = 0; nt < 4; ++nt)
#pragma unroll
            for (int r = 0; r < 8; ++r) c[mt][nt][r] = 0.0f;

    const uint4 zero4 = make_uint4(0u, 0u, 0u, 0u);

    for (int k0 = 0; k0 < K; k0 += 32) {
        // ---- stage A tile [64][32] : 256 16B-chunks, 2 per thread ----
#pragma unroll
        for (int it = 0; it < 2; ++it) {
            int ch = tid + it * 128;
            int r  = ch >> 2, kc = (ch & 3) * 8;
            unsigned short* dst = &Asub[r * 32 + kc];
            if (GUARD) {
                int gr = m0 + r;
                if (gr < M) copy16_g2l(A + (long long)gr * lda + (k0 + kc), dst);
                else        *(uint4*)dst = zero4;
            } else {
                copy16_g2l(A + (long long)(m0 + r) * lda + (k0 + kc), dst);
            }
        }
        // ---- stage B tile [128][32] : 512 16B-chunks, 4 per thread ----
#pragma unroll
        for (int it = 0; it < 4; ++it) {
            int ch = tid + it * 128;
            int n  = ch >> 2, kc = (ch & 3) * 8;
            unsigned short* dst = &Bsub[n * 32 + kc];
            if (GUARD) {
                int gn = n0 + n;
                if (gn < N) copy16_g2l(Bg + (long long)gn * ldb + (k0 + kc), dst);
                else        *(uint4*)dst = zero4;
            } else {
                copy16_g2l(Bg + (long long)(n0 + n) * ldb + (k0 + kc), dst);
            }
        }
        wait_async_lds();
        __syncthreads();

        // ---- fragments (ISA 7.12.2 layouts; lower as ds_load_b128) ----
        AFrag a[2], bfr[4];
#pragma unroll
        for (int mt = 0; mt < 2; ++mt) {
            const unsigned short* ap = &Asub[(wm + mt * 16 + mrow) * 32];
#pragma unroll
            for (int j = 0; j < 8; ++j) {
                int k = ((j >> 2) << 4) + (khalf << 3) + ((j & 3) << 1);
                a[mt].u[j] = *(const unsigned int*)(ap + k);
            }
        }
#pragma unroll
        for (int nt = 0; nt < 4; ++nt) {
            const unsigned short* bp = &Bsub[(wn + nt * 16 + mrow) * 32];
#pragma unroll
            for (int j = 0; j < 8; ++j) {
                int k = ((j >> 2) << 4) + (khalf << 3) + ((j & 3) << 1);
                bfr[nt].u[j] = *(const unsigned int*)(bp + k);
            }
        }
#pragma unroll
        for (int mt = 0; mt < 2; ++mt)
#pragma unroll
            for (int nt = 0; nt < 4; ++nt)
                c[mt][nt] = __builtin_amdgcn_wmma_f32_16x16x32_bf16(
                    false, a[mt].v, false, bfr[nt].v,
                    (short)0, c[mt][nt], false, false);
        __syncthreads();
    }

    // ---- epilogue ----
#pragma unroll
    for (int mt = 0; mt < 2; ++mt)
#pragma unroll
        for (int nt = 0; nt < 4; ++nt)
#pragma unroll
            for (int r = 0; r < 8; ++r) {
                int row = m0 + wm + mt * 16 + r + (khalf << 3);
                int col = n0 + wn + nt * 16 + mrow;
                if (!GUARD || (row < M && col < N)) {
                    float v = c[mt][nt][r];
                    if (bias) v += bias[col];
                    if (act == 1)  // exact GELU
                        v = 0.5f * v * (1.0f + erff(v * 0.70710678118654752f));
                    long long o = coff + (long long)row * ldc + col;
                    if (res)   v += res[o];
                    if (outf)  outf[o] = v;
                    if (outb)  outb[o] = f2bf(v);
                    if (outbT) outbT[coff + (long long)col * ldcT + row] = f2bf(v);
                }
            }
}

// ---------------- LayerNorm over D=384, wave per row ----------------
__global__ __launch_bounds__(256)
void ln_kernel(const float* __restrict__ in, const float* __restrict__ sc,
               const float* __restrict__ bi,
               float* __restrict__ outf, unsigned short* __restrict__ outb,
               int Mrows)
{
    int row  = blockIdx.x * 8 + (threadIdx.x >> 5);
    if (row >= Mrows) return;
    int lane = threadIdx.x & 31;
    const float* p = in + (long long)row * DIM_;
    float x[12], s = 0.0f;
#pragma unroll
    for (int i = 0; i < 12; ++i) { x[i] = p[lane + i * 32]; s += x[i]; }
#pragma unroll
    for (int off = 16; off; off >>= 1) s += __shfl_xor(s, off, 32);
    float mean = s * (1.0f / DIM_);
    float var = 0.0f;
#pragma unroll
    for (int i = 0; i < 12; ++i) { float d = x[i] - mean; var += d * d; }
#pragma unroll
    for (int off = 16; off; off >>= 1) var += __shfl_xor(var, off, 32);
    float inv = rsqrtf(var * (1.0f / DIM_) + 1e-5f);
#pragma unroll
    for (int i = 0; i < 12; ++i) {
        int d = lane + i * 32;
        float y = (x[i] - mean) * inv * sc[d] + bi[d];
        long long o = (long long)row * DIM_ + d;
        if (outf) outf[o] = y;
        if (outb) outb[o] = f2bf(y);
    }
}

// ---------------- RoPE: fp32 qk[M,768] -> bf16 q,k [M,384] ----------------
__global__ __launch_bounds__(256)
void rope_kernel(const float* __restrict__ qk, const float* __restrict__ freqs,
                 unsigned short* __restrict__ qb, unsigned short* __restrict__ kb)
{
    int idx = blockIdx.x * 256 + threadIdx.x;       // MTOK * HEADS * NF
    if (idx >= MTOK * HEADS_ * NF_) return;
    int f   = idx & (NF_ - 1);
    int h   = (idx >> 5) % HEADS_;
    int tok = idx / (HEADS_ * NF_);
    int n   = tok & (NTOK - 1);
    float pi = (float)(n >> 5);                     // grid row
    float pj = (float)(n & 31);                     // grid col
    float f0 = freqs[(h * NF_ + f) * 2 + 0];
    float f1 = freqs[(h * NF_ + f) * 2 + 1];
    float th = f0 * pi + f1 * pj;
    float cth = cosf(th), sth = sinf(th);

    const float* qrow = qk + (long long)tok * 768 + h * DH_;
    const float* krow = qrow + DIM_;
    long long ob = (long long)tok * DIM_ + h * DH_;
    float qx = qrow[f], qy = qrow[NF_ + f];
    qb[ob + f]       = f2bf(qx * cth - qy * sth);
    qb[ob + NF_ + f] = f2bf(qx * sth + qy * cth);
    float kx = krow[f], ky = krow[NF_ + f];
    kb[ob + f]       = f2bf(kx * cth - ky * sth);
    kb[ob + NF_ + f] = f2bf(kx * sth + ky * cth);
}

// ---------------- softmax over 1024 keys, wave per row ----------------
__global__ __launch_bounds__(256)
void softmax_kernel(const float* __restrict__ S, unsigned short* __restrict__ P,
                    float scale)
{
    int row  = blockIdx.x * 8 + (threadIdx.x >> 5);   // 0 .. HEADS*NTOK-1
    int lane = threadIdx.x & 31;
    const float* p = S + (long long)row * NTOK;
    float v[32], m = -3.0e38f;
#pragma unroll
    for (int i = 0; i < 32; ++i) { v[i] = p[lane + i * 32] * scale; m = fmaxf(m, v[i]); }
#pragma unroll
    for (int off = 16; off; off >>= 1) m = fmaxf(m, __shfl_xor(m, off, 32));
    float s = 0.0f;
#pragma unroll
    for (int i = 0; i < 32; ++i) { v[i] = expf(v[i] - m); s += v[i]; }
#pragma unroll
    for (int off = 16; off; off >>= 1) s += __shfl_xor(s, off, 32);
    float inv = 1.0f / s;
    unsigned short* q = P + (long long)row * NTOK;
#pragma unroll
    for (int i = 0; i < 32; ++i) q[lane + i * 32] = f2bf(v[i] * inv);
}

// ---------------- patchify: x[8,3,512,512] -> bf16 Ap[8192,768] ----------------
__global__ __launch_bounds__(256)
void patchify_kernel(const float* __restrict__ x, unsigned short* __restrict__ Ap)
{
    long long idx = (long long)blockIdx.x * 256 + threadIdx.x;
    if (idx >= (long long)MTOK * KPATCH) return;
    int col = (int)(idx % KPATCH);
    int row = (int)(idx / KPATCH);
    int ch = col % 3;
    int pp = col / 3;
    int p0 = pp >> 4, p1 = pp & 15;
    int gj = row & 31;
    int fi = (row >> 5) & 31;
    int b  = row >> 10;
    long long xi = (((long long)(b * 3 + ch) * 512) + fi * 16 + p0) * 512 + gj * 16 + p1;
    Ap[idx] = f2bf(x[xi]);
}

// ---------------- mean pool over tokens ----------------
__global__ __launch_bounds__(256)
void meanpool_kernel(const float* __restrict__ tln, unsigned short* __restrict__ pooled)
{
    int idx = blockIdx.x * 256 + threadIdx.x;      // B_IMG * DIM
    if (idx >= B_IMG * DIM_) return;
    int d = idx % DIM_, b = idx / DIM_;
    const float* p = tln + (long long)b * NTOK * DIM_ + d;
    float s = 0.0f;
    for (int n = 0; n < NTOK; ++n) s += p[(long long)n * DIM_];
    pooled[idx] = f2bf(s * (1.0f / NTOK));
}

// ---------------- fp32 [L,K,N] -> bf16 transposed [L,N,K] ----------------
__global__ __launch_bounds__(256)
void convT_kernel(const float* __restrict__ in, unsigned short* __restrict__ out,
                  int K, int N, long long total)
{
    long long idx = (long long)blockIdx.x * 256 + threadIdx.x;
    if (idx >= total) return;
    int n = (int)(idx % N);
    long long r = idx / N;
    int k = (int)(r % K);
    long long l = r / K;
    out[(l * N + n) * (long long)K + k] = f2bf(in[idx]);
}

// =========================== host launcher ===========================
extern "C" void kernel_launch(void* const* d_in, const int* in_sizes, int n_in,
                              void* d_out, int out_size, void* d_ws, size_t ws_size,
                              hipStream_t stream)
{
    (void)in_sizes; (void)n_in; (void)out_size; (void)ws_size;
    const float* x          = (const float*)d_in[0];
    const float* patch_w    = (const float*)d_in[1];
    const float* patch_b    = (const float*)d_in[2];
    const float* patch_ln_s = (const float*)d_in[3];
    const float* patch_ln_b = (const float*)d_in[4];
    const float* freqs      = (const float*)d_in[5];
    const float* attn_ln_s  = (const float*)d_in[6];
    const float* attn_ln_b  = (const float*)d_in[7];
    const float* qk_w       = (const float*)d_in[8];
    const float* v_w        = (const float*)d_in[9];
    const float* out_w      = (const float*)d_in[10];
    const float* out_b      = (const float*)d_in[11];
    const float* ff_ln_s    = (const float*)d_in[12];
    const float* ff_ln_b    = (const float*)d_in[13];
    const float* ff_w1      = (const float*)d_in[14];
    const float* ff_b1      = (const float*)d_in[15];
    const float* ff_w2      = (const float*)d_in[16];
    const float* ff_b2      = (const float*)d_in[17];
    const float* final_ln_s = (const float*)d_in[18];
    const float* final_ln_b = (const float*)d_in[19];
    const float* head_w     = (const float*)d_in[20];
    const float* head_b     = (const float*)d_in[21];
    float* out = (float*)d_out;

    // ---- workspace carve (256B aligned) ----
    char* ws = (char*)d_ws;
    size_t off = 0;
    auto alloc = [&](size_t bytes) -> void* {
        void* p = ws + off;
        off = (off + bytes + 255) & ~(size_t)255;
        return p;
    };
    const long long nWp  = (long long)KPATCH * DIM_;
    const long long nWqk = 8LL * DIM_ * 768;
    const long long nWv  = 8LL * DIM_ * DIM_;
    const long long nWo  = 8LL * DIM_ * DIM_;
    const long long nW1  = 8LL * DIM_ * MLP_;
    const long long nW2  = 8LL * MLP_ * DIM_;
    const long long nWh  = (long long)DIM_ * NCLS_;

    // transposed bf16 weights [N,K]
    unsigned short* WpT  = (unsigned short*)alloc(nWp  * 2);
    unsigned short* WqkT = (unsigned short*)alloc(nWqk * 2);
    unsigned short* WvT  = (unsigned short*)alloc(nWv  * 2);
    unsigned short* WoT  = (unsigned short*)alloc(nWo  * 2);
    unsigned short* W1T  = (unsigned short*)alloc(nW1  * 2);
    unsigned short* W2T  = (unsigned short*)alloc(nW2  * 2);
    unsigned short* WhT  = (unsigned short*)alloc(nWh  * 2);

    unsigned short* Ap  = (unsigned short*)alloc((long long)MTOK * KPATCH * 2);
    float*          t   = (float*)alloc((long long)MTOK * DIM_ * 4);
    float*          tmp = (float*)alloc((long long)MTOK * 768 * 4);
    unsigned short* hbf = (unsigned short*)alloc((long long)MTOK * DIM_ * 2);
    unsigned short* qbf = (unsigned short*)alloc((long long)MTOK * DIM_ * 2);
    unsigned short* kbf = (unsigned short*)alloc((long long)MTOK * DIM_ * 2);
    unsigned short* vtb = (unsigned short*)alloc((long long)DIM_ * MTOK * 2); // V^T [384][8192]
    unsigned short* obf = (unsigned short*)alloc((long long)MTOK * DIM_ * 2);
    unsigned short* gbf = (unsigned short*)alloc((long long)MTOK * MLP_ * 2);
    float*          S   = (float*)alloc((long long)HEADS_ * NTOK * NTOK * 4);
    unsigned short* Pp  = (unsigned short*)alloc((long long)HEADS_ * NTOK * NTOK * 2);
    float*          tln = (float*)alloc((long long)MTOK * DIM_ * 4);
    unsigned short* pooled = (unsigned short*)alloc((long long)B_IMG * DIM_ * 2);

    auto convT = [&](const float* src, unsigned short* dst, int K, int N, long long tot) {
        convT_kernel<<<(int)((tot + 255) / 256), 256, 0, stream>>>(src, dst, K, N, tot);
    };
    auto gemm = [&](const unsigned short* A, const unsigned short* Bw,
                    const float* bias, const float* res,
                    float* outf, unsigned short* outb, unsigned short* outbT,
                    int M, int N, int K, int lda, int ldb, int ldc, int ldcT,
                    int act, int nz,
                    long long sA, long long sB, long long sC) {
        dim3 g((N + 127) / 128, (M + 63) / 64, nz);
        bool guard = (M % 64) != 0 || (N % 128) != 0;
        if (guard)
            wmma_gemm_kernel<true><<<g, 128, 0, stream>>>(A, Bw, bias, res, outf, outb, outbT,
                                                          M, N, K, lda, ldb, ldc, ldcT,
                                                          act, sA, sB, sC);
        else
            wmma_gemm_kernel<false><<<g, 128, 0, stream>>>(A, Bw, bias, res, outf, outb, outbT,
                                                           M, N, K, lda, ldb, ldc, ldcT,
                                                           act, sA, sB, sC);
    };

    // ---- convert weights to transposed bf16 [N,K] ----
    convT(patch_w, WpT, KPATCH, DIM_, nWp);
    convT(qk_w,  WqkT, DIM_, 768,  nWqk);
    convT(v_w,   WvT,  DIM_, DIM_, nWv);
    convT(out_w, WoT,  DIM_, DIM_, nWo);
    convT(ff_w1, W1T,  DIM_, MLP_, nW1);
    convT(ff_w2, W2T,  MLP_, DIM_, nW2);
    convT(head_w, WhT, DIM_, NCLS_, nWh);

    // ---- patch embed ----
    {
        long long n = (long long)MTOK * KPATCH;
        patchify_kernel<<<(int)((n + 255) / 256), 256, 0, stream>>>(x, Ap);
    }
    gemm(Ap, WpT, patch_b, nullptr, tmp, nullptr, nullptr,
         MTOK, DIM_, KPATCH, KPATCH, KPATCH, DIM_, 0, 0, 1, 0, 0, 0);
    ln_kernel<<<MTOK / 8, 256, 0, stream>>>(tmp, patch_ln_s, patch_ln_b,
                                            t, nullptr, MTOK);

    const float scale = 0.125f;   // DH^-0.5
    const long long bStride = (long long)NTOK * DIM_;   // per-image stride in q/k/o

    for (int l = 0; l < 8; ++l) {
        // attn LN -> bf16
        ln_kernel<<<MTOK / 8, 256, 0, stream>>>(t, attn_ln_s + l * DIM_,
                                                attn_ln_b + l * DIM_,
                                                nullptr, hbf, MTOK);
        // qk projection: [M,384] x [384,768] -> fp32 tmp
        gemm(hbf, WqkT + (long long)l * DIM_ * 768, nullptr, nullptr, tmp, nullptr, nullptr,
             MTOK, 768, DIM_, DIM_, DIM_, 768, 0, 0, 1, 0, 0, 0);
        // v projection -> bf16 transposed V^T [384][8192]
        gemm(hbf, WvT + (long long)l * DIM_ * DIM_, nullptr, nullptr, nullptr, nullptr, vtb,
             MTOK, DIM_, DIM_, DIM_, DIM_, 0, MTOK, 0, 1, 0, 0, 0);
        // RoPE -> qbf, kbf
        rope_kernel<<<(MTOK * HEADS_ * NF_) / 256, 256, 0, stream>>>(tmp, freqs, qbf, kbf);

        // attention per image, batched over heads (blockIdx.z)
        for (int b = 0; b < B_IMG; ++b) {
            // S = Q K^T   (K rows are [token, dh] == B[N,K] layout)
            gemm(qbf + b * bStride, kbf + b * bStride, nullptr, nullptr, S, nullptr, nullptr,
                 NTOK, NTOK, DH_, DIM_, DIM_, NTOK, 0, 0, HEADS_,
                 DH_, DH_, (long long)NTOK * NTOK);
            softmax_kernel<<<(HEADS_ * NTOK) / 8, 256, 0, stream>>>(S, Pp, scale);
            // O = P V : B = V^T slice [dh][ntok] per (b,h)  (N=64 -> guarded path)
            gemm(Pp, vtb + (long long)b * NTOK, nullptr, nullptr,
                 nullptr, obf + b * bStride, nullptr,
                 NTOK, DH_, NTOK, NTOK, MTOK, DIM_, 0, 0, HEADS_,
                 (long long)NTOK * NTOK, (long long)DH_ * MTOK, DH_);
        }
        // out projection + bias + residual -> t
        gemm(obf, WoT + (long long)l * DIM_ * DIM_, out_b + l * DIM_, t, t, nullptr, nullptr,
             MTOK, DIM_, DIM_, DIM_, DIM_, DIM_, 0, 0, 1, 0, 0, 0);

        // ff LN -> bf16
        ln_kernel<<<MTOK / 8, 256, 0, stream>>>(t, ff_ln_s + l * DIM_,
                                                ff_ln_b + l * DIM_,
                                                nullptr, hbf, MTOK);
        // FF1 + gelu -> bf16 gbf
        gemm(hbf, W1T + (long long)l * DIM_ * MLP_, ff_b1 + l * MLP_, nullptr,
             nullptr, gbf, nullptr,
             MTOK, MLP_, DIM_, DIM_, DIM_, MLP_, 0, 1, 1, 0, 0, 0);
        // FF2 + bias + residual -> t
        gemm(gbf, W2T + (long long)l * MLP_ * DIM_, ff_b2 + l * DIM_, t, t, nullptr, nullptr,
             MTOK, DIM_, MLP_, MLP_, MLP_, DIM_, 0, 0, 1, 0, 0, 0);
    }

    // ---- final LN, mean pool, classifier head ----
    ln_kernel<<<MTOK / 8, 256, 0, stream>>>(t, final_ln_s, final_ln_b,
                                            tln, nullptr, MTOK);
    meanpool_kernel<<<(B_IMG * DIM_ + 255) / 256, 256, 0, stream>>>(tln, pooled);
    gemm(pooled, WhT, head_b, nullptr, out, nullptr, nullptr,
         B_IMG, NCLS_, DIM_, DIM_, DIM_, NCLS_, 0, 0, 1, 0, 0, 0);
}
